// MultiHeadAttention_6339371729167
// MI455X (gfx1250) — matrix operand order repaired and checked by reading, not compiled, not run on hardware
//
#include <hip/hip_runtime.h>

// ---------------------------------------------------------------------------
// MI455X / gfx1250 implementation of the dual-source MHA block.
// GEMM-shaped math on v_wmma_f32_16x16x32_f16; A-tiles staged to LDS with
// global_load_async_to_lds_b128 (ASYNCcnt); K/V pre-swizzled into f16
// WMMA-fragment order so attention inner loops are load+wmma only.
// ---------------------------------------------------------------------------

typedef __attribute__((ext_vector_type(16))) _Float16 v16h;
typedef __attribute__((ext_vector_type(4)))  _Float16 v4h;
typedef __attribute__((ext_vector_type(8)))  float    v8f;

// Problem constants
#define BB 4
#define NN 1024          // H*W
#define CC 256
#define HD 32            // head dim
#define MM (BB * NN)     // 4096 rows for the projection GEMMs

// ---------------------------------------------------------------------------
// 1. Positional encoding. pe[n][c]; div_term[2j] == div_term[2j+1] so the
//    effective frequency index is (c & 63) within each 64-channel block.
// ---------------------------------------------------------------------------
__global__ void pe_kernel(float* __restrict__ pe) {
    int i = blockIdx.x * blockDim.x + threadIdx.x;
    if (i >= NN * CC) return;
    int c = i & 255;
    int n = i >> 8;
    int y = n >> 5, x = n & 31;
    float pos, cc;
    if (c < 128) { pos = (float)(y + 1); cc = (float)(c & 63); }          // py
    else         { pos = (float)(x + 1); cc = (float)((c - 128) & 63); }  // px
    float div = __powf(10000.0f, cc * (1.0f / 64.0f));
    pe[i] = __sinf(pos / div);
}

// ---------------------------------------------------------------------------
// 2. Repack a 256x256 fp32 weight (W[k][n], n contiguous) into f16
//    B-fragment order: tile (kt: 32 K's, nt: 16 N's), 32 lanes x 16 halfs.
//    B order used here (same as A): lane -> n = lane%16, elem e -> k = e + 16*(lane/16).
// ---------------------------------------------------------------------------
__global__ void pack_w(const float* __restrict__ W, _Float16* __restrict__ Wp) {
    int t = blockIdx.x * blockDim.x + threadIdx.x;   // 0..4095
    if (t >= 128 * 32) return;
    int lane = t & 31;
    int tile = t >> 5;            // kt*16 + nt
    int kt = tile >> 4, nt = tile & 15;
    int n  = nt * 16 + (lane & 15);
    int kb = kt * 32 + ((lane & 16) ? 16 : 0);
    _Float16 tmp[16];
#pragma unroll
    for (int e = 0; e < 16; ++e) tmp[e] = (_Float16)W[(kb + e) * CC + n];
    *(v16h*)(Wp + (size_t)t * 16) = *(v16h*)tmp;     // 32B aligned store
}

// ---------------------------------------------------------------------------
// 3. GEMM: out[4096x256] = (A (+ pe)) @ Wp + bias.  4 waves / block, each
//    wave owns one 16x16 tile; 8 K-steps of wmma_f32_16x16x32_f16.
//    The 16x256 f32 A-tile is staged to LDS with async b128 loads, then PE
//    is added in place once, so the K-loop is pure LDS->cvt->wmma.
// ---------------------------------------------------------------------------
__global__ __launch_bounds__(128)
void gemm_kernel(const float* __restrict__ A, const float* __restrict__ pe,
                 const _Float16* __restrict__ Wp, const float* __restrict__ bias,
                 float* __restrict__ out) {
    __shared__ __align__(16) float As[16 * CC];      // 16 KB

    const int tid   = threadIdx.x;
    const int lane  = tid & 31;
    const int wave  = tid >> 5;
    const int mt    = blockIdx.x;                  // 16-row tile (0..255)
    const int nt    = blockIdx.y * 4 + wave;       // 16-col tile (0..15)
    const int kbase = (lane & 16) ? 16 : 0;

    // ---- async-stage A tile: 1024 x 16B, 8 per thread ----
    {
        const float* gsrc = A + (size_t)mt * 16 * CC;
        unsigned ldsbase = (unsigned)(uintptr_t)(&As[0]);
        for (int i = tid; i < 1024; i += 128) {
            unsigned           lo = ldsbase + (unsigned)i * 16u;
            unsigned long long ga = (unsigned long long)(uintptr_t)(gsrc + i * 4);
            asm volatile("global_load_async_to_lds_b128 %0, %1, off"
                         :: "v"(lo), "v"(ga) : "memory");
        }
        asm volatile("s_wait_asynccnt 0x0" ::: "memory");
        __syncthreads();
        if (pe) {   // fuse PE add in place (rows mt*16.. stay in one pe block)
            const float* pesrc = pe + ((size_t)((mt * 16) & (NN - 1))) * CC;
            for (int i = tid; i < 1024; i += 128) {
                float4  pv = *(const float4*)(pesrc + i * 4);
                float4* dp = (float4*)(&As[i * 4]);
                float4  av = *dp;
                av.x += pv.x; av.y += pv.y; av.z += pv.z; av.w += pv.w;
                *dp = av;
            }
            __syncthreads();
        }
    }

    // ---- K loop ----
    const float* arow = As + (lane & 15) * CC;
    v8f acc = {};
    for (int kt = 0; kt < 8; ++kt) {
        if (kt < 7)  // prefetch next weight tile (global_prefetch_b8)
            __builtin_prefetch(Wp + ((size_t)((kt + 1) * 16 + nt) * 32) * 16, 0, 1);

        const float* p = arow + kt * 32 + kbase;
        float ab[16];
        *(float4*)(ab + 0)  = *(const float4*)(p);
        *(float4*)(ab + 4)  = *(const float4*)(p + 4);
        *(float4*)(ab + 8)  = *(const float4*)(p + 8);
        *(float4*)(ab + 12) = *(const float4*)(p + 12);
        v16h a;
#pragma unroll
        for (int e = 0; e < 16; ++e) a[e] = (_Float16)ab[e];

        v16h b = *(const v16h*)(Wp + ((size_t)(kt * 16 + nt) * 32 + lane) * 16);
        acc = __builtin_amdgcn_wmma_f32_16x16x32_f16(false, a, false, b,
                                                     (short)0, acc, false, false);
    }

    // D layout: vgpr r, lane -> (M = r + 8*(lane/16), N = lane%16)
    const int n  = nt * 16 + (lane & 15);
    const float bv = bias ? bias[n] : 0.0f;
#pragma unroll
    for (int r = 0; r < 8; ++r) {
        int mg = mt * 16 + r + 8 * (lane >> 4);
        out[(size_t)mg * CC + n] = acc[r] + bv;
    }
}

// ---------------------------------------------------------------------------
// 4. Head fold: qs[m][c] = q[m][c] + q[m][c+128]   (8 heads -> 4)
// ---------------------------------------------------------------------------
__global__ void fold_kernel(const float* __restrict__ q, float* __restrict__ qs) {
    int i = blockIdx.x * blockDim.x + threadIdx.x;
    if (i >= MM * 128) return;
    int c = i & 127;
    int m = i >> 7;
    qs[i] = q[(size_t)m * CC + c] + q[(size_t)m * CC + 128 + c];
}

// ---------------------------------------------------------------------------
// 5. 3x3 SAME conv on k[...,128:192] -> kc[...,0:64]; kc[...,64:128] copies
//    k[...,192:256].  conv_w is HWIO (3,3,64,64).
// ---------------------------------------------------------------------------
__global__ void conv_kernel(const float* __restrict__ k, const float* __restrict__ cw,
                            const float* __restrict__ cb, float* __restrict__ kc) {
    int idx = blockIdx.x * blockDim.x + threadIdx.x;
    if (idx >= BB * NN * 128) return;
    int co = idx & 127;
    int x  = (idx >> 7) & 31;
    int y  = (idx >> 12) & 31;
    int b  = idx >> 17;
    float acc;
    if (co < 64) {
        acc = cb[co];
        for (int dy = -1; dy <= 1; ++dy) {
            int yy = y + dy; if (yy < 0 || yy > 31) continue;
            for (int dx = -1; dx <= 1; ++dx) {
                int xx = x + dx; if (xx < 0 || xx > 31) continue;
                const float* src = k + ((size_t)((b * 32 + yy) * 32 + xx)) * CC + 128;
                const float* w   = cw + (size_t)(((dy + 1) * 3 + (dx + 1)) * 64) * 64 + co;
                float s = 0.0f;
#pragma unroll 8
                for (int ci = 0; ci < 64; ++ci) s += src[ci] * w[ci * 64];
                acc += s;
            }
        }
    } else {
        acc = k[((size_t)((b * 32 + y) * 32 + x)) * CC + 128 + co]; // 192+(co-64)
    }
    kc[idx] = acc;
}

// ---------------------------------------------------------------------------
// 6a. Pre-swizzle K side into f16 B-fragment order (K^T fragments).
//     Kp[((b*8+h)*64 + j)*32 + lane][16]: key = j*16 + lane%16,
//     k-dim elem e -> e + 16*(lane/16).  Heads 0-3 read kconv, 4-7 read k.
// ---------------------------------------------------------------------------
__global__ void kpack_kernel(const float* __restrict__ kb, const float* __restrict__ kcv,
                             _Float16* __restrict__ Kp) {
    int t = blockIdx.x * blockDim.x + threadIdx.x;   // 0..65535
    if (t >= BB * 8 * 64 * 32) return;
    int lane = t & 31;
    int j    = (t >> 5) & 63;
    int h    = (t >> 11) & 7;
    int b    = t >> 14;
    int key  = j * 16 + (lane & 15);
    int kd   = (lane & 16) ? 16 : 0;
    const float* src = (h < 4)
        ? kcv + ((size_t)(b * NN + key)) * 128 + h * HD + kd
        : kb  + ((size_t)(b * NN + key)) * CC  + (h - 4) * HD + kd;
    float f[16];
    *(float4*)(f + 0)  = *(const float4*)(src + 0);
    *(float4*)(f + 4)  = *(const float4*)(src + 4);
    *(float4*)(f + 8)  = *(const float4*)(src + 8);
    *(float4*)(f + 12) = *(const float4*)(src + 12);
    _Float16 tmp[16];
#pragma unroll
    for (int e = 0; e < 16; ++e) tmp[e] = (_Float16)f[e];
    *(v16h*)(Kp + (size_t)t * 16) = *(v16h*)tmp;
}

// ---------------------------------------------------------------------------
// 6b. Pre-swizzle V into f16 B-fragment order for P@V.
//     Vp[(((b*8+h)*32 + jj)*2 + which)*32 + lane][16]:
//     dim = lane%16 + which*16, key elem e -> jj*32 + e + 16*(lane/16).
// ---------------------------------------------------------------------------
__global__ void vpack_kernel(const float* __restrict__ v, _Float16* __restrict__ Vp) {
    int t = blockIdx.x * blockDim.x + threadIdx.x;   // 0..65535
    if (t >= BB * 8 * 32 * 2 * 32) return;
    int lane  = t & 31;
    int which = (t >> 5) & 1;
    int jj    = (t >> 6) & 31;
    int h     = (t >> 11) & 7;
    int b     = t >> 14;
    int dim   = (lane & 15) + which * 16;
    int key0  = jj * 32 + ((lane & 16) ? 16 : 0);
    const float* src = v + ((size_t)(b * NN + key0)) * CC + h * HD + dim;
    _Float16 tmp[16];
#pragma unroll
    for (int e = 0; e < 16; ++e) tmp[e] = (_Float16)src[(size_t)e * CC];
    *(v16h*)(Vp + (size_t)t * 16) = *(v16h*)tmp;
}

// ---------------------------------------------------------------------------
// 7. Attention: one wave per (b, head, 16-query tile).
//    Heads 0-3: q2s ("co");  heads 4-7: q1s ("self").  K/V come pre-packed.
//    LDS: 16x1024 f32 logits (64KB) + 16x1024 f16 probs (32KB) = 96KB.
// ---------------------------------------------------------------------------
__global__ __launch_bounds__(32)
void attn_kernel(const float* __restrict__ q1s, const float* __restrict__ q2s,
                 const _Float16* __restrict__ Kp, const _Float16* __restrict__ Vp,
                 float* __restrict__ out) {
    extern __shared__ char smem[];
    float*    lg = (float*)smem;                         // [16][1024]
    _Float16* P  = (_Float16*)(smem + 16 * 1024 * 4);    // [16][1024]

    const int lane = threadIdx.x;
    const int qt   = blockIdx.x;   // 0..63
    const int h    = blockIdx.y;   // 0..7 (global head)
    const int b    = blockIdx.z;

    const float* qs = (h < 4) ? q2s : q1s;
    const int    hh = (h < 4) ? h : (h - 4);
    const int kbase = (lane & 16) ? 16 : 0;

    // ---- Q A-fragment (16x32; d == WMMA K) ----
    const int mrow = qt * 16 + (lane & 15);
    const float* qrow = qs + ((size_t)(b * NN + mrow)) * 128 + hh * HD;
    float qb[16];
    *(float4*)(qb + 0)  = *(const float4*)(qrow + kbase);
    *(float4*)(qb + 4)  = *(const float4*)(qrow + kbase + 4);
    *(float4*)(qb + 8)  = *(const float4*)(qrow + kbase + 8);
    *(float4*)(qb + 12) = *(const float4*)(qrow + kbase + 12);
    v16h aq;
#pragma unroll
    for (int e = 0; e < 16; ++e) aq[e] = (_Float16)qb[e];

    const float invscale = 0.17677669529663687f;  // 1/sqrt(32)

    // ---- Phase 1: logits = (Q K^T)/sqrt(d); one v16h load + one WMMA/chunk ----
    const _Float16* kpb = Kp + ((size_t)((b * 8 + h) * 64) * 32) * 16;
    for (int j = 0; j < 64; ++j) {
        v16h bk = *(const v16h*)(kpb + ((size_t)j * 32 + lane) * 16);
        v8f d = {};
        d = __builtin_amdgcn_wmma_f32_16x16x32_f16(false, aq, false, bk,
                                                   (short)0, d, false, false);
        const int col = j * 16 + (lane & 15);
#pragma unroll
        for (int r = 0; r < 8; ++r)
            lg[(r + 8 * (lane >> 4)) * NN + col] = d[r] * invscale;
    }
    __syncthreads();

    // ---- Phase 2: softmax; lane -> row = lane%16, half of the keys ----
    {
        const int row = lane & 15;
        const int h0  = (lane >> 4) * 512;
        const float4* lg4 = (const float4*)(lg + row * NN + h0);
        float mx = -1e30f;
        for (int i = 0; i < 128; ++i) {
            float4 t = lg4[i];
            mx = fmaxf(mx, fmaxf(fmaxf(t.x, t.y), fmaxf(t.z, t.w)));
        }
        mx = fmaxf(mx, __shfl_xor(mx, 16, 32));
        float s = 0.0f;
        for (int i = 0; i < 128; ++i) {
            float4 t = lg4[i];
            s += __expf(t.x - mx) + __expf(t.y - mx) + __expf(t.z - mx) + __expf(t.w - mx);
        }
        s += __shfl_xor(s, 16, 32);
        const float inv = 1.0f / s;
        for (int i = 0; i < 128; ++i) {
            float4 t = lg4[i];
            v4h pv;
            pv[0] = (_Float16)(__expf(t.x - mx) * inv);
            pv[1] = (_Float16)(__expf(t.y - mx) * inv);
            pv[2] = (_Float16)(__expf(t.z - mx) * inv);
            pv[3] = (_Float16)(__expf(t.w - mx) * inv);
            *(v4h*)(P + row * NN + h0 + 4 * i) = pv;
        }
    }
    __syncthreads();

    // ---- Phase 3: out = P @ V; two v16h loads + two WMMAs per 32-key chunk ----
    const _Float16* vpb = Vp + ((size_t)((b * 8 + h) * 32) * 2 * 32) * 16;
    v8f o0 = {}, o1 = {};
    for (int jj = 0; jj < 32; ++jj) {
        const _Float16* pp = P + (lane & 15) * NN + jj * 32;
        v16h pa;
#pragma unroll
        for (int e = 0; e < 16; ++e)
            pa[e] = pp[e + kbase];

        v16h vb0 = *(const v16h*)(vpb + (((size_t)jj * 2 + 0) * 32 + lane) * 16);
        v16h vb1 = *(const v16h*)(vpb + (((size_t)jj * 2 + 1) * 32 + lane) * 16);
        o0 = __builtin_amdgcn_wmma_f32_16x16x32_f16(false, pa, false, vb0,
                                                    (short)0, o0, false, false);
        o1 = __builtin_amdgcn_wmma_f32_16x16x32_f16(false, pa, false, vb1,
                                                    (short)0, o1, false, false);
    }

    // ---- Store: out[b, n, h*32 + dim] ----
#pragma unroll
    for (int r = 0; r < 8; ++r) {
        const int mg = qt * 16 + r + 8 * (lane >> 4);
        float* orow = out + ((size_t)(b * NN + mg)) * CC + h * HD;
        orow[(lane & 15)]      = o0[r];
        orow[16 + (lane & 15)] = o1[r];
    }
}

// ---------------------------------------------------------------------------
// Host-side orchestration
// ---------------------------------------------------------------------------
extern "C" void kernel_launch(void* const* d_in, const int* in_sizes, int n_in,
                              void* d_out, int out_size, void* d_ws, size_t ws_size,
                              hipStream_t stream) {
    (void)in_sizes; (void)n_in; (void)out_size;
    const float* src1   = (const float*)d_in[0];
    const float* src2   = (const float*)d_in[1];
    const float* Wq1    = (const float*)d_in[2];
    const float* bq1    = (const float*)d_in[3];
    const float* Wq2    = (const float*)d_in[4];
    const float* bq2    = (const float*)d_in[5];
    const float* Wk     = (const float*)d_in[6];
    const float* bk     = (const float*)d_in[7];
    const float* Wv     = (const float*)d_in[8];
    const float* bv     = (const float*)d_in[9];
    const float* conv_w = (const float*)d_in[10];
    const float* conv_b = (const float*)d_in[11];
    const float* Wo     = (const float*)d_in[12];
    const float* bo     = (const float*)d_in[13];

    char* ws = (char*)d_ws;
    const size_t MB = 1024 * 1024;
    float*    pe   = (float*)(ws + 0 * MB);          // 1 MB
    float*    q1   = (float*)(ws + 1 * MB);          // 4 MB
    float*    q2   = (float*)(ws + 5 * MB);          // 4 MB
    float*    kb   = (float*)(ws + 9 * MB);          // 4 MB
    float*    vb   = (float*)(ws + 13 * MB);         // 4 MB
    float*    q1s  = (float*)(ws + 17 * MB);         // 2 MB
    float*    q2s  = (float*)(ws + 19 * MB);         // 2 MB
    float*    kcv  = (float*)(ws + 21 * MB);         // 2 MB
    float*    aout = (float*)(ws + 23 * MB);         // 4 MB
    _Float16* Wp0  = (_Float16*)(ws + 27 * MB);      // 5 x 128 KB packed weights
    _Float16* Wp1  = Wp0 + 65536;
    _Float16* Wp2  = Wp1 + 65536;
    _Float16* Wp3  = Wp2 + 65536;
    _Float16* Wp4  = Wp3 + 65536;
    _Float16* Kp   = (_Float16*)(ws + 28 * MB);      // 2 MB packed K fragments
    _Float16* Vp   = (_Float16*)(ws + 30 * MB);      // 2 MB packed V fragments
    if (ws_size < 32 * MB) return;  // workspace too small; nothing safe to do

    // 1. positional encoding
    pe_kernel<<<1024, 256, 0, stream>>>(pe);

    // 2. pack weights to WMMA B-fragment order (f16)
    pack_w<<<16, 256, 0, stream>>>(Wq1, Wp0);
    pack_w<<<16, 256, 0, stream>>>(Wq2, Wp1);
    pack_w<<<16, 256, 0, stream>>>(Wk,  Wp2);
    pack_w<<<16, 256, 0, stream>>>(Wv,  Wp3);
    pack_w<<<16, 256, 0, stream>>>(Wo,  Wp4);

    // 3. projections (pe fused; A staged via async-to-LDS)
    dim3 ggrid(256, 4);
    gemm_kernel<<<ggrid, 128, 0, stream>>>(src1, pe, Wp0, bq1, q1);
    gemm_kernel<<<ggrid, 128, 0, stream>>>(src2, pe, Wp1, bq2, q2);
    gemm_kernel<<<ggrid, 128, 0, stream>>>(src1, pe, Wp2, bk,  kb);
    gemm_kernel<<<ggrid, 128, 0, stream>>>(src1, pe, Wp3, bv,  vb);

    // 4. head folds
    fold_kernel<<<2048, 256, 0, stream>>>(q1, q1s);
    fold_kernel<<<2048, 256, 0, stream>>>(q2, q2s);

    // 5. conv branch of K
    conv_kernel<<<2048, 256, 0, stream>>>(kb, conv_w, conv_b, kcv);

    // 6. pre-swizzle K and V into WMMA fragment order
    kpack_kernel<<<256, 256, 0, stream>>>(kb, kcv, Kp);
    vpack_kernel<<<256, 256, 0, stream>>>(vb, Vp);

    // 7. attention (96 KB dynamic LDS per workgroup; WGP has 320 KB)
    dim3 agrid(64, 8, 4);
    size_t shmem = 16 * 1024 * 4 + 16 * 1024 * 2;
    attn_kernel<<<agrid, 32, shmem, stream>>>(q1s, q2s, Kp, Vp, aout);

    // 8. output projection -> d_out
    gemm_kernel<<<ggrid, 128, 0, stream>>>(aout, nullptr, Wp4, bo, (float*)d_out);
}
